// PointNet2_82317343195434
// MI455X (gfx1250) — matrix operand order repaired and checked by reading, not compiled, not run on hardware
//
#include <hip/hip_runtime.h>
#include <hip/hip_bf16.h>

typedef __attribute__((ext_vector_type(16))) __bf16 v16bf;
typedef __attribute__((ext_vector_type(8)))  float  v8f;

#define N_PTS 16384
#define KNN_K 10
#define BN_EPS 1e-5f

// fp32 -> bf16 (RNE), as raw 16-bit pattern
__device__ __forceinline__ unsigned short f2bfu(float a) {
  unsigned ua = __builtin_bit_cast(unsigned, a);
  ua = ua + 0x7FFFu + ((ua >> 16) & 1u);
  return (unsigned short)(ua >> 16);
}

// ---------------------------------------------------------------------------
// 1) Preprocess: coords4 = [x,y,z,batch], x0 = [nc(3), feat]
// ---------------------------------------------------------------------------
__global__ void build_x0_kernel(const float* __restrict__ pc,
                                float* __restrict__ coords4,
                                float* __restrict__ x0, int n) {
  int i = blockIdx.x * blockDim.x + threadIdx.x;
  if (i >= n) return;
  float cx = pc[i * 5 + 0], cy = pc[i * 5 + 1], cz = pc[i * 5 + 2];
  float bt = pc[i * 5 + 3], ft = pc[i * 5 + 4];
  coords4[i * 4 + 0] = cx; coords4[i * 4 + 1] = cy;
  coords4[i * 4 + 2] = cz; coords4[i * 4 + 3] = bt;
  const float inv = 1.0f / 384.0f; // SS/2 = 384
  x0[i * 4 + 0] = (cx - 384.0f) * inv;
  x0[i * 4 + 1] = (cy - 384.0f) * inv;
  x0[i * 4 + 2] = (cz - 384.0f) * inv;
  x0[i * 4 + 3] = ft;
}

// ---------------------------------------------------------------------------
// 2) Weight pre-convert: Wt[c*Cin + k] = bf16(W[k*Cout + c])  (transposed)
// ---------------------------------------------------------------------------
__global__ void cvtW_kernel(const float* __restrict__ W,
                            unsigned short* __restrict__ Wt,
                            int Cin, int Cout) {
  int t = blockIdx.x * blockDim.x + threadIdx.x;
  if (t >= Cin * Cout) return;
  int k = t / Cout, c = t % Cout;
  Wt[(size_t)c * Cin + k] = f2bfu(W[t]);
}

// ---------------------------------------------------------------------------
// 3) Fused streaming KNN: no NxN matrix, LDS position tiles, register top-10
// ---------------------------------------------------------------------------
__global__ void knn_kernel(const float4* __restrict__ coords4,
                           int* __restrict__ knn, int n) {
  __shared__ float4 tile[256];
  int i = blockIdx.x * 256 + threadIdx.x; // one query per thread, grid exact
  float4 p = coords4[i];
  float bestd[KNN_K];
  int   besti[KNN_K];
#pragma unroll
  for (int k = 0; k < KNN_K; ++k) { bestd[k] = __builtin_inff(); besti[k] = 0; }

  for (int jt = 0; jt < n; jt += 256) {
    tile[threadIdx.x] = coords4[jt + threadIdx.x];
    __syncthreads();
#pragma unroll 4
    for (int jj = 0; jj < 256; ++jj) {
      float4 q = tile[jj];
      int j = jt + jj;
      float dx = p.x - q.x, dy = p.y - q.y, dz = p.z - q.z;
      float d2 = dx * dx + dy * dy + dz * dz;
      bool bad = (q.w != p.w) || (j == i);
      if (!bad && d2 < bestd[KNN_K - 1]) {
        float dd = d2; int ji = j;      // swap-chain insertion, fully unrolled
#pragma unroll
        for (int t = 0; t < KNN_K; ++t) {
          if (dd < bestd[t]) {
            float td = bestd[t]; int ti = besti[t];
            bestd[t] = dd; besti[t] = ji;
            dd = td; ji = ti;
          }
        }
      }
    }
    __syncthreads();
  }
#pragma unroll
  for (int k = 0; k < KNN_K; ++k) knn[(size_t)i * KNN_K + k] = besti[k];
}

// ---------------------------------------------------------------------------
// 4) Aggregation: y[i] = x[i] + sum_k x[idx[i][k]]  (fp32 + bf16 sidecar)
// ---------------------------------------------------------------------------
__global__ void agg_kernel(const float* __restrict__ X,
                           const int* __restrict__ idx,
                           float* __restrict__ Y,
                           unsigned short* __restrict__ Yb, int n, int C) {
  size_t t = (size_t)blockIdx.x * blockDim.x + threadIdx.x;
  if (t >= (size_t)n * C) return;
  int i = (int)(t / C), c = (int)(t % C);
  float s = X[(size_t)i * C + c];
  const int* ip = idx + (size_t)i * KNN_K;
#pragma unroll
  for (int k = 0; k < KNN_K; ++k) s += X[(size_t)ip[k] * C + c];
  Y[t]  = s;
  Yb[t] = f2bfu(s);
}

// ---------------------------------------------------------------------------
// 5) WMMA GEMM: Y[N,Cout] = act(Xb[N,Cin](bf16) @ Wt(bf16,transposed) + b)
//    Block = 128 threads = 4 waves; block tile = 64 rows x (NSUB*16) cols.
//    Each wave: 16 rows, one A fragment shared across NSUB B fragments ->
//    NSUB back-to-back WMMAs per K-chunk, NSUB accumulators.
//    Conversion-free hot loop: X/W staging are pure b128 global->LDS copies
//    (Wt is pre-transposed bf16, K-major per column); fragments are
//    ds_load_b128 groups. OOB W *columns* are clamped (their outputs are
//    never stored); only K is zero-padded (tail layers Cin=4/16).
//    A layout: lanes 0-15 K 0..15, lanes 16-31 K 16..31 (row = lane&15).
//    B layout: lanes 0-15 K 0..15, lanes 16-31 K 16..31 (col = lane&15).
// ---------------------------------------------------------------------------
template <int NSUB>
__global__ void gemm_bias_act(const unsigned short* __restrict__ Xb,
                              const unsigned short* __restrict__ Wt,
                              const float* __restrict__ Bias,
                              float* __restrict__ Y,
                              int Cin, int Cout, int leaky) {
  // rows x 32 bf16 -> 16 dwords per row; cols x 32 k -> 16 dwords per col
  __shared__ __attribute__((aligned(16))) unsigned sX[64 * 16];
  __shared__ __attribute__((aligned(16))) unsigned sWt[NSUB * 16 * 16];

  int tid  = threadIdx.x;
  int wave = tid >> 5, lane = tid & 31;
  int half = lane >> 4, m = lane & 15;
  int rowBase = blockIdx.x * 64;
  int colBase = blockIdx.y * (NSUB * 16);

  // X staging: each thread owns 16 consecutive k of one row of the tile
  int srow = tid >> 1;                 // 0..63
  int skh  = (tid & 1) << 4;           // 0 or 16
  const unsigned short* xs = Xb + (size_t)(rowBase + srow) * Cin + skh;
  unsigned* xdst = &sX[srow * 16 + (skh >> 1)];   // 8 dwords

  // W staging (same shape as X): 2 threads per column, 16 bf16 each
  int wcol = tid >> 1;                 // 0..63 (only < NSUB*16 participate)
  bool wAct = wcol < NSUB * 16;
  int cg = colBase + (wAct ? wcol : 0);
  if (cg >= Cout) cg = Cout - 1;       // clamp: outputs of these cols dropped
  const unsigned short* wsrc = Wt + (size_t)cg * Cin + skh;
  unsigned* wdst = &sWt[(wAct ? wcol : 0) * 16 + (skh >> 1)];

  v8f acc[NSUB];
#pragma unroll
  for (int s = 0; s < NSUB; ++s) acc[s] = (v8f){};

  for (int kb = 0; kb < Cin; kb += 32) {
    if (kb + 32 <= Cin) {              // uniform fast path: pure b128 copies
      const uint4* s4 = (const uint4*)(xs + kb);
      uint4 d0 = s4[0], d1 = s4[1];
      *(uint4*)(xdst)     = d0;
      *(uint4*)(xdst + 4) = d1;
      if (wAct) {
        const uint4* w4 = (const uint4*)(wsrc + kb);
        uint4 e0 = w4[0], e1 = w4[1];
        *(uint4*)(wdst)     = e0;
        *(uint4*)(wdst + 4) = e1;
      }
    } else {                           // tail chunk (Cin = 4 or 16 only)
#pragma unroll
      for (int q = 0; q < 8; ++q) {
        int kg = kb + skh + 2 * q;
        unsigned u0 = xs[kb + 2 * q];        // in-bounds of N*1024 buffer
        unsigned u1 = xs[kb + 2 * q + 1];
        u0 = (kg     < Cin) ? u0 : 0u;       // branchless zero-pad
        u1 = (kg + 1 < Cin) ? u1 : 0u;
        xdst[q] = u0 | (u1 << 16);
      }
      if (wAct) {
#pragma unroll
        for (int q = 0; q < 8; ++q) {
          int kg = kb + skh + 2 * q;
          unsigned u0 = wsrc[(kg     < Cin) ? (kb + 2 * q)     : 0];
          unsigned u1 = wsrc[(kg + 1 < Cin) ? (kb + 2 * q + 1) : 0];
          u0 = (kg     < Cin) ? u0 : 0u;
          u1 = (kg + 1 < Cin) ? u1 : 0u;
          wdst[q] = u0 | (u1 << 16);
        }
      }
    }
    __syncthreads();

    if (kb + 32 < Cin) {
      __builtin_prefetch(&xs[kb + 32], 0, 0);
      __builtin_prefetch(&wsrc[kb + 32], 0, 0);
    }

    // ---- fragments from LDS (unguarded, 16B-aligned dword groups) ----
    union { v16bf v; unsigned u[8]; } a, b;
    int rl = wave * 16 + m;
#pragma unroll
    for (int j = 0; j < 4; ++j) a.u[j]     = sX[rl * 16 + half * 8 + j];
#pragma unroll
    for (int j = 0; j < 4; ++j) a.u[4 + j] = sX[rl * 16 + half * 8 + 4 + j];

#pragma unroll
    for (int s = 0; s < NSUB; ++s) {   // NSUB back-to-back WMMAs, shared A
#pragma unroll
      for (int j = 0; j < 8; ++j) b.u[j] = sWt[(s * 16 + m) * 16 + half * 8 + j];
      acc[s] = __builtin_amdgcn_wmma_f32_16x16x32_bf16(
          false, a.v, false, b.v, (short)0, acc[s], false, false);
    }
    __syncthreads();
  }

#pragma unroll
  for (int s = 0; s < NSUB; ++s) {
    int col = colBase + s * 16 + m;
    float bias = Bias[(col < Cout) ? col : 0];   // branchless, clamped
#pragma unroll
    for (int v = 0; v < 8; ++v) {      // C/D layout: row = base + half*8 + v
      int r = rowBase + wave * 16 + half * 8 + v;
      float y = acc[s][v] + bias;
      if (leaky) y = (y > 0.0f) ? y : 0.33f * y;
      if (col < Cout) Y[(size_t)r * Cout + col] = y;
    }
  }
}

// ---------------------------------------------------------------------------
// 6) Batchnorm stats per column: stats[c]=mean, stats[C+c]=rsqrt(var+eps)
// ---------------------------------------------------------------------------
__global__ void colstats_kernel(const float* __restrict__ Y,
                                float* __restrict__ stats, int n, int C) {
  __shared__ float ss[256], sq[256];
  int col = blockIdx.x;
  float s = 0.0f, s2 = 0.0f;
  for (int r = threadIdx.x; r < n; r += blockDim.x) {
    float v = Y[(size_t)r * C + col];
    s += v; s2 += v * v;
  }
  ss[threadIdx.x] = s; sq[threadIdx.x] = s2;
  __syncthreads();
  for (int off = 128; off > 0; off >>= 1) {
    if (threadIdx.x < off) {
      ss[threadIdx.x] += ss[threadIdx.x + off];
      sq[threadIdx.x] += sq[threadIdx.x + off];
    }
    __syncthreads();
  }
  if (threadIdx.x == 0) {
    float mean = ss[0] / (float)n;
    float var  = sq[0] / (float)n - mean * mean;
    stats[col]     = mean;
    stats[C + col] = rsqrtf(var + BN_EPS);
  }
}

// normalize in place; also emit bf16 sidecar for the next GEMM
__global__ void bn_apply_kernel(float* __restrict__ Y,
                                unsigned short* __restrict__ Yb,
                                const float* __restrict__ stats, int n, int C) {
  size_t t = (size_t)blockIdx.x * blockDim.x + threadIdx.x;
  if (t >= (size_t)n * C) return;
  int col = (int)(t % C);
  float v = (Y[t] - stats[col]) * stats[C + col];
  Y[t]  = v;
  Yb[t] = f2bfu(v);
}

// ---------------------------------------------------------------------------
// Host orchestration
// ---------------------------------------------------------------------------
static inline size_t align256(size_t x) { return (x + 255) & ~(size_t)255; }

extern "C" void kernel_launch(void* const* d_in, const int* in_sizes, int n_in,
                              void* d_out, int out_size, void* d_ws, size_t ws_size,
                              hipStream_t stream) {
  const int n = N_PTS;
  const float* pc = (const float*)d_in[0];

  // ---- map param pointers (handle insertion-order vs key-sorted pytree) ----
  const float* Wp[13]; const float* bp[13];
  if (n_in >= 3 && in_sizes[1] == 1024 * 22) {
    // sorted: final first, then gin1, gin2, gin3, mlp3
    Wp[12] = (const float*)d_in[1]; bp[12] = (const float*)d_in[2];
    for (int l = 0; l < 12; ++l) {
      Wp[l] = (const float*)d_in[3 + 2 * l];
      bp[l] = (const float*)d_in[4 + 2 * l];
    }
  } else {
    // insertion order: gin1, gin2, gin3, mlp3, final
    for (int l = 0; l < 13; ++l) {
      Wp[l] = (const float*)d_in[1 + 2 * l];
      bp[l] = (const float*)d_in[2 + 2 * l];
    }
  }
  static const int CIN[13]  = {4, 16, 16, 32, 64, 64, 64, 128, 128, 128, 256, 256, 1024};
  static const int COUT[13] = {16, 16, 32, 64, 64, 64, 128, 128, 128, 256, 256, 1024, 22};

  // ---- workspace carve ----
  char* p = (char*)d_ws;
  float*          coords4 = (float*)p;          p += align256((size_t)n * 4 * sizeof(float));
  int*            knn     = (int*)p;            p += align256((size_t)n * KNN_K * sizeof(int));
  float*          bufA    = (float*)p;          p += align256((size_t)n * 1024 * sizeof(float));
  float*          bufB    = (float*)p;          p += align256((size_t)n * 1024 * sizeof(float));
  unsigned short* bufXb   = (unsigned short*)p; p += align256((size_t)n * 1024 * sizeof(unsigned short));
  float*          stats   = (float*)p;          p += align256(2 * 1024 * sizeof(float));
  unsigned short* wtArena = (unsigned short*)p; // sum(Cin*Cout) bf16
  (void)ws_size;

  unsigned short* Wt[13];
  {
    size_t off = 0;
    for (int l = 0; l < 13; ++l) {
      Wt[l] = wtArena + off;
      off += (size_t)CIN[l] * COUT[l];
    }
  }

  // ---- weight pre-convert (tiny; once per call) ----
  for (int l = 0; l < 13; ++l) {
    int ne = CIN[l] * COUT[l];
    cvtW_kernel<<<(ne + 255) / 256, 256, 0, stream>>>(Wp[l], Wt[l], CIN[l], COUT[l]);
  }

  // ---- preprocess + KNN ----
  build_x0_kernel<<<(n + 255) / 256, 256, 0, stream>>>(pc, coords4, bufA, n);
  knn_kernel<<<n / 256, 256, 0, stream>>>((const float4*)coords4, knn, n);

  auto run_gemm = [&](float* Y, int l, int leaky) {
    int Cout = COUT[l];
    int nsub = (Cout + 15) / 16; if (nsub > 4) nsub = 4;   // 16->1,22/32->2,64+->4
    dim3 grid(n / 64, (Cout + nsub * 16 - 1) / (nsub * 16));
    if (nsub == 1)
      gemm_bias_act<1><<<grid, 128, 0, stream>>>(bufXb, Wt[l], bp[l], Y, CIN[l], Cout, leaky);
    else if (nsub == 2)
      gemm_bias_act<2><<<grid, 128, 0, stream>>>(bufXb, Wt[l], bp[l], Y, CIN[l], Cout, leaky);
    else
      gemm_bias_act<4><<<grid, 128, 0, stream>>>(bufXb, Wt[l], bp[l], Y, CIN[l], Cout, leaky);
  };
  auto run_bn = [&](float* Y, int C) {
    colstats_kernel<<<C, 256, 0, stream>>>(Y, stats, n, C);
    bn_apply_kernel<<<(int)(((size_t)n * C + 255) / 256), 256, 0, stream>>>(Y, bufXb, stats, n, C);
  };
  auto run_agg = [&](const float* X, float* Y, int C) {
    agg_kernel<<<(int)(((size_t)n * C + 255) / 256), 256, 0, stream>>>(X, knn, Y, bufXb, n, C);
  };

  // ---- GIN stage 1: agg(x0) then 4->16->16->32 ----
  run_agg(bufA, bufB, 4);
  run_gemm(bufA, 0, 1); run_bn(bufA, 16);
  run_gemm(bufB, 1, 1); run_bn(bufB, 16);
  run_gemm(bufA, 2, 1); run_bn(bufA, 32);

  // ---- GIN stage 2: agg then 32->64->64->64 ----
  run_agg(bufA, bufB, 32);
  run_gemm(bufA, 3, 1); run_bn(bufA, 64);
  run_gemm(bufB, 4, 1); run_bn(bufB, 64);
  run_gemm(bufA, 5, 1); run_bn(bufA, 64);

  // ---- GIN stage 3: agg then 64->128->128->128 ----
  run_agg(bufA, bufB, 64);
  run_gemm(bufA, 6, 1); run_bn(bufA, 128);
  run_gemm(bufB, 7, 1); run_bn(bufB, 128);
  run_gemm(bufA, 8, 1); run_bn(bufA, 128);

  // ---- mlp3: 128->256->256->1024 (no agg) ----
  run_gemm(bufB, 9, 1);  run_bn(bufB, 256);
  run_gemm(bufA, 10, 1); run_bn(bufA, 256);
  run_gemm(bufB, 11, 1); run_bn(bufB, 1024);

  // ---- final: 1024 -> 22, plain linear, straight to d_out ----
  run_gemm((float*)d_out, 12, 0);
  (void)out_size;
}
